// Attention_2010044694851
// MI455X (gfx1250) — matrix operand order repaired and checked by reading, not compiled, not run on hardware
//
#include <hip/hip_runtime.h>
#include <hip/hip_bf16.h>

// ---------------------------------------------------------------------------
// Causal attention:  qkv = x@Wqkv^T ; S = qk^T*scale (causal) ; P = softmax(S)
//                    o = P@v ; out = o@Wout^T + b
// All matmuls via V_WMMA_F32_16X16X32_BF16 (bf16 in, f32 accum), wave32.
// ---------------------------------------------------------------------------

#define BATCH   4
#define SEQ     2048
#define DIN     1024
#define DINNER  512
#define DOUT    1024
#define ROWS    (BATCH * SEQ)          // 8192

typedef __attribute__((ext_vector_type(16))) __bf16 bf16x16;
typedef __attribute__((ext_vector_type(8)))  __bf16 bf16x8;
typedef __attribute__((ext_vector_type(8)))  float  f32x8;

static __device__ __forceinline__
f32x8 wmma_bf16(bf16x16 a, bf16x16 b, f32x8 c) {
    return __builtin_amdgcn_wmma_f32_16x16x32_bf16(
        /*neg_a*/false, a, /*neg_b*/false, b,
        /*c_mod*/(short)0, c, /*reuse_a*/false, /*reuse_b*/false);
}

// A fragment (16x32, M x K), source row-major with leading dim ld.
// lane&15 = row within tile; lane>>4 selects K sub-block.
static __device__ __forceinline__
bf16x16 load_a(const __bf16* base, int row, int ld, int k0, int lane) {
    int half = lane >> 4;
    const __bf16* p = base + (size_t)row * ld + k0 + half * 16;
    bf16x8 lo = *(const bf16x8*)(p);
    bf16x8 hi = *(const bf16x8*)(p + 8);
    bf16x16 a;
#pragma unroll
    for (int i = 0; i < 8; ++i) { a[i] = lo[i]; a[i + 8] = hi[i]; }
    return a;
}

// B fragment (32x16, K x N) where B[k][n] = src[n][k]  (src row-major, ld).
// lane&15 = column n of the tile; per-lane 16 contiguous K values (32B load).
static __device__ __forceinline__
bf16x16 load_b(const __bf16* src, int nrow, int ld, int k0, int lane) {
    int half = lane >> 4;
    return *(const bf16x16*)(src + (size_t)nrow * ld + k0 + half * 16);
}

// ---------------------------------------------------------------------------
__global__ void k_cvt_bf16(const float* __restrict__ in,
                           __bf16* __restrict__ out, int n) {
    for (int i = blockIdx.x * blockDim.x + threadIdx.x; i < n;
         i += gridDim.x * blockDim.x)
        out[i] = (__bf16)in[i];
}

// ---------------------------------------------------------------------------
// qkv = x @ Wqkv^T.  One wave -> 16x64 tile of qkv (K = 1024).
// Columns 0..511 -> q, 512..1023 -> k (row-major), 1024..1535 -> vT[d][seq].
__global__ void k_qkv(const __bf16* __restrict__ xb,
                      const __bf16* __restrict__ wb,
                      __bf16* __restrict__ qb, __bf16* __restrict__ kb,
                      __bf16* __restrict__ vT) {
    int wave = blockIdx.x * (blockDim.x >> 5) + (threadIdx.x >> 5);
    int lane = threadIdx.x & 31;
    int rt = wave / 24, ct = wave % 24;          // 512 row tiles x 24 col tiles
    int rbase = rt * 16, n0 = ct * 64;
    int half = lane >> 4, nl = lane & 15;

    f32x8 acc[4] = {};
#pragma unroll 4
    for (int k0 = 0; k0 < DIN; k0 += 32) {
        bf16x16 a = load_a(xb, rbase + nl, DIN, k0, lane);
#pragma unroll
        for (int t = 0; t < 4; ++t) {
            bf16x16 b = load_b(wb, n0 + t * 16 + nl, DIN, k0, lane);
            acc[t] = wmma_bf16(a, b, acc[t]);
        }
    }
    int region = n0 >> 9;                         // 0=q 1=k 2=v
    int batch = rbase >> 11;
#pragma unroll
    for (int t = 0; t < 4; ++t) {
        int col = n0 + t * 16 + nl;
        if (region < 2) {
            __bf16* dst = region ? kb : qb;
            int e = col - region * DINNER;
#pragma unroll
            for (int r = 0; r < 8; ++r) {
                int grow = rbase + r + 8 * half;
                dst[(size_t)grow * DINNER + e] = (__bf16)acc[t][r];
            }
        } else {                                  // v: transposed store
            int d = col - 2 * DINNER;
            int seq0 = (rbase & (SEQ - 1)) + 8 * half;
            bf16x8 pk;
#pragma unroll
            for (int r = 0; r < 8; ++r) pk[r] = (__bf16)acc[t][r];
            *(bf16x8*)(vT + ((size_t)batch * DINNER + d) * SEQ + seq0) = pk;
        }
    }
}

// ---------------------------------------------------------------------------
// S[b][i][j] = scale * q_i . k_j  (masked).  One wave per 16-row query tile.
// q A-fragments (16 of them) preloaded and reused across all key tiles.
__global__ void k_scores(const __bf16* __restrict__ qb,
                         const __bf16* __restrict__ kb,
                         float* __restrict__ S) {
    int wave = blockIdx.x * (blockDim.x >> 5) + (threadIdx.x >> 5);
    int lane = threadIdx.x & 31;
    int batch = wave >> 7, qt = wave & 127;
    int qbase = qt * 16;
    int half = lane >> 4, nl = lane & 15;
    const __bf16* qB = qb + (size_t)batch * SEQ * DINNER;
    const __bf16* kB = kb + (size_t)batch * SEQ * DINNER;
    float* Sb = S + (size_t)batch * SEQ * SEQ;
    const float scale = 0.044194173824159216f;    // 512^-0.5

    bf16x16 qa[16];
#pragma unroll
    for (int c = 0; c < 16; ++c)
        qa[c] = load_a(qB, qbase + nl, DINNER, c * 32, lane);

    for (int kt = 0; kt <= qt; ++kt) {
        int kbase = kt * 16;
        __builtin_prefetch(kB + (size_t)(kbase + 16) * DINNER, 0, 0);
        f32x8 c = {};
#pragma unroll
        for (int kc = 0; kc < 16; ++kc) {
            bf16x16 b = load_b(kB, kbase + nl, DINNER, kc * 32, lane);
            c = wmma_bf16(qa[kc], b, c);
        }
        int j = kbase + nl;
#pragma unroll
        for (int r = 0; r < 8; ++r) {
            int i = qbase + r + 8 * half;
            float s = c[r] * scale;
            if (j > i) s = -3.0e38f;
            Sb[(size_t)i * SEQ + j] = s;
        }
    }
}

// ---------------------------------------------------------------------------
// Row softmax over the causally-defined prefix; writes P (bf16), zero-padded
// up to the next multiple of 32 so the PV GEMM has no tail.
__global__ void k_softmax(const float* __restrict__ S,
                          __bf16* __restrict__ P) {
    int wave = blockIdx.x * (blockDim.x >> 5) + (threadIdx.x >> 5);
    int lane = threadIdx.x & 31;
    int row = wave;                                // 0..8191
    int i = row & (SEQ - 1);
    int L = ((i >> 4) + 1) * 16;
    int Lz = (L + 31) & ~31;
    const float* srow = S + (size_t)row * SEQ;
    __bf16* prow = P + (size_t)row * SEQ;

    float m = -3.4e38f;
    for (int j = lane; j < L; j += 32) m = fmaxf(m, srow[j]);
#pragma unroll
    for (int d = 16; d >= 1; d >>= 1) m = fmaxf(m, __shfl_xor(m, d, 32));

    float sum = 0.f;
    for (int j = lane; j < L; j += 32) sum += __expf(srow[j] - m);
#pragma unroll
    for (int d = 16; d >= 1; d >>= 1) sum += __shfl_xor(sum, d, 32);
    float inv = 1.f / sum;

    for (int j = lane; j < Lz; j += 32) {
        float sv = (j < L) ? srow[j] : -3.4e38f;
        prow[j] = (__bf16)(__expf(sv - m) * inv);
    }
}

// ---------------------------------------------------------------------------
// o = P @ v  using vT; K-loop truncated at the causal limit of the tile.
// One wave -> 16x64 tile of o.
__global__ void k_pv(const __bf16* __restrict__ P,
                     const __bf16* __restrict__ vT,
                     __bf16* __restrict__ ob) {
    int wave = blockIdx.x * (blockDim.x >> 5) + (threadIdx.x >> 5);
    int lane = threadIdx.x & 31;
    int batch = wave >> 10;
    int rem = wave & 1023;
    int qt = rem >> 3, ct = rem & 7;
    int qbase = qt * 16, n0 = ct * 64;
    int half = lane >> 4, nl = lane & 15;
    const __bf16* Pb = P + (size_t)batch * SEQ * SEQ;
    const __bf16* vB = vT + (size_t)batch * DINNER * SEQ;
    int Kz = (((qt + 1) * 16) + 31) & ~31;

    f32x8 acc[4] = {};
    for (int k0 = 0; k0 < Kz; k0 += 32) {
        bf16x16 a = load_a(Pb, qbase + nl, SEQ, k0, lane);
#pragma unroll
        for (int t = 0; t < 4; ++t) {
            bf16x16 b = load_b(vB, n0 + t * 16 + nl, SEQ, k0, lane);
            acc[t] = wmma_bf16(a, b, acc[t]);
        }
    }
#pragma unroll
    for (int t = 0; t < 4; ++t) {
        int d = n0 + t * 16 + nl;
#pragma unroll
        for (int r = 0; r < 8; ++r) {
            int grow = batch * SEQ + qbase + r + 8 * half;
            ob[(size_t)grow * DINNER + d] = (__bf16)acc[t][r];
        }
    }
}

// ---------------------------------------------------------------------------
// out = o @ Wout^T + b.  One wave -> 16x64 tile, f32 output.
__global__ void k_outproj(const __bf16* __restrict__ ob,
                          const __bf16* __restrict__ wb,
                          const float* __restrict__ bias,
                          float* __restrict__ out) {
    int wave = blockIdx.x * (blockDim.x >> 5) + (threadIdx.x >> 5);
    int lane = threadIdx.x & 31;
    int rt = wave >> 4, ct = wave & 15;           // 512 row tiles x 16 col
    int rbase = rt * 16, n0 = ct * 64;
    int half = lane >> 4, nl = lane & 15;

    f32x8 acc[4] = {};
#pragma unroll 4
    for (int k0 = 0; k0 < DINNER; k0 += 32) {
        bf16x16 a = load_a(ob, rbase + nl, DINNER, k0, lane);
#pragma unroll
        for (int t = 0; t < 4; ++t) {
            bf16x16 b = load_b(wb, n0 + t * 16 + nl, DINNER, k0, lane);
            acc[t] = wmma_bf16(a, b, acc[t]);
        }
    }
#pragma unroll
    for (int t = 0; t < 4; ++t) {
        int col = n0 + t * 16 + nl;
        float bv = bias[col];
#pragma unroll
        for (int r = 0; r < 8; ++r) {
            int grow = rbase + r + 8 * half;
            out[(size_t)grow * DOUT + col] = acc[t][r] + bv;
        }
    }
}

// ---------------------------------------------------------------------------
extern "C" void kernel_launch(void* const* d_in, const int* in_sizes, int n_in,
                              void* d_out, int out_size, void* d_ws, size_t ws_size,
                              hipStream_t stream) {
    const float* x     = (const float*)d_in[0];
    const float* w_qkv = (const float*)d_in[1];
    const float* w_out = (const float*)d_in[2];
    const float* b_out = (const float*)d_in[3];
    float* out = (float*)d_out;

    // Workspace layout (bytes) ~148 MB total; intermediates live in L2 (192MB).
    char* ws = (char*)d_ws;
    __bf16* xb    = (__bf16*)(ws);                          // 16 MiB
    __bf16* wqkvb = (__bf16*)(ws + 16777216);               //  3 MiB
    __bf16* woutb = (__bf16*)(ws + 19922944);               //  1 MiB
    __bf16* qb    = (__bf16*)(ws + 20971520);               //  8 MiB
    __bf16* kb    = (__bf16*)(ws + 29360128);               //  8 MiB
    __bf16* vT    = (__bf16*)(ws + 37748736);               //  8 MiB
    float*  S     = (float*) (ws + 46137344);               // 64 MiB
    __bf16* P     = (__bf16*)(ws + 113246208);              // 32 MiB
    __bf16* ob    = (__bf16*)(ws + 146800640);              //  8 MiB

    // 0) f32 -> bf16 conversions
    k_cvt_bf16<<<2048, 256, 0, stream>>>(x,     xb,    ROWS * DIN);
    k_cvt_bf16<<<1024, 256, 0, stream>>>(w_qkv, wqkvb, 3 * DINNER * DIN);
    k_cvt_bf16<<<512,  256, 0, stream>>>(w_out, woutb, DOUT * DINNER);

    // 1) qkv projection: 512 row tiles * 24 col tiles, 4 waves/block
    k_qkv<<<(512 * 24) / 4, 128, 0, stream>>>(xb, wqkvb, qb, kb, vT);

    // 2) attention scores: 4 batches * 128 query tiles
    k_scores<<<512 / 4, 128, 0, stream>>>(qb, kb, S);

    // 3) row softmax: 8192 rows, one wave each
    k_softmax<<<ROWS / 4, 128, 0, stream>>>(S, P);

    // 4) P @ v: 4 * 128 * 8 tiles
    k_pv<<<4096 / 4, 128, 0, stream>>>(P, vT, ob);

    // 5) output projection: 512 * 16 tiles
    k_outproj<<<8192 / 4, 128, 0, stream>>>(ob, woutb, b_out, out);

    (void)in_sizes; (void)n_in; (void)out_size; (void)ws_size;
}
